// OnlineCenterLoss_82927228551475
// MI455X (gfx1250) — matrix operand order repaired and checked by reading, not compiled, not run on hardware
//
#include <hip/hip_runtime.h>

// OnlineCenterLoss for MI455X (gfx1250, wave32, WMMA), bf16x3 split-precision GEMM.
//
// loss_val[i,c] = 1 + (2*e_i.c_c - |c_c|^2) - (2*e_i.c_tgt - |c_tgt|^2)
// (|e_i|^2 cancels). Pipeline:
//   k0: split E and C into bf16 hi/lo pairs (x = hi + lo, |lo| <= 2^-9 |x|)
//   k1: c2[c] = |center_c|^2                       (f32 exact, one wave/center)
//   k2: t[i]  = 2*e_i.c_{tgt[i]} - c2[tgt[i]]      (f32 exact, one wave/row)
//   k3: fused GEMM: dot ~= ehi.chi + ehi.clo + elo.chi via v_wmma_f32_16x16x32_bf16
//       (error ~2^-18 relative; 8x FLOP/instr vs the f32 16x16x4 shape, 3 terms
//        => ~2.7x fewer matrix-pipe cycles) + mask + per-WG reduction
//   k4: deterministic final tree reduction -> scalar

typedef float  v8f  __attribute__((ext_vector_type(8)));
typedef __bf16 v16bf __attribute__((ext_vector_type(16)));
typedef __bf16 v8bf  __attribute__((ext_vector_type(8)));
typedef __bf16 v4bf  __attribute__((ext_vector_type(4)));

#define N_ROWS 16384
#define N_CTR  4096
#define DIM    128
#define LAMBD  1.0f

#define BM 64            // embedding rows per workgroup tile
#define BN 256           // center cols per workgroup tile
#define LDSB (DIM + 8)   // padded bf16 row stride (136 elems = 272B): 4-bank skew, 16B aligned
#define NBLK ((N_ROWS / BM) * (N_CTR / BN))  // 4096 workgroups in k3

// ---------------------------------------------------------------- kernel 0
__global__ __launch_bounds__(256) void split_bf16_kernel(const float* __restrict__ src,
                                                         __bf16* __restrict__ hi,
                                                         __bf16* __restrict__ lo) {
  const int idx = blockIdx.x * 256 + threadIdx.x;
  float4 v = reinterpret_cast<const float4*>(src)[idx];
  float f[4] = {v.x, v.y, v.z, v.w};
  v4bf h, l;
#pragma unroll
  for (int j = 0; j < 4; ++j) {
    __bf16 hj = (__bf16)f[j];
    h[j] = hj;
    l[j] = (__bf16)(f[j] - (float)hj);
  }
  *reinterpret_cast<v4bf*>(hi + idx * 4) = h;
  *reinterpret_cast<v4bf*>(lo + idx * 4) = l;
}

// ---------------------------------------------------------------- kernel 1
__global__ __launch_bounds__(256) void c2_kernel(const float* __restrict__ centers,
                                                 float* __restrict__ c2) {
  const int wave = threadIdx.x >> 5;
  const int lane = threadIdx.x & 31;
  const int row  = blockIdx.x * 8 + wave;
  float4 v = reinterpret_cast<const float4*>(centers + row * DIM)[lane];
  float s = v.x * v.x + v.y * v.y + v.z * v.z + v.w * v.w;
#pragma unroll
  for (int m = 16; m >= 1; m >>= 1) s += __shfl_xor(s, m, 32);
  if (lane == 0) c2[row] = s;
}

// ---------------------------------------------------------------- kernel 2
__global__ __launch_bounds__(256) void t_kernel(const float* __restrict__ emb,
                                                const int* __restrict__ tgt,
                                                const float* __restrict__ centers,
                                                const float* __restrict__ c2,
                                                float* __restrict__ t) {
  const int wave = threadIdx.x >> 5;
  const int lane = threadIdx.x & 31;
  const int row  = blockIdx.x * 8 + wave;
  const int tg   = tgt[row];
  float4 e = reinterpret_cast<const float4*>(emb + row * DIM)[lane];
  float4 c = reinterpret_cast<const float4*>(centers + tg * DIM)[lane];
  float s = e.x * c.x + e.y * c.y + e.z * c.z + e.w * c.w;
#pragma unroll
  for (int m = 16; m >= 1; m >>= 1) s += __shfl_xor(s, m, 32);
  if (lane == 0) t[row] = 2.0f * s - c2[tg];
}

// ---------------------------------------------------------------- kernel 3
__global__ __launch_bounds__(256) void gemm_loss_kernel(
    const __bf16* __restrict__ Ehi, const __bf16* __restrict__ Elo,
    const __bf16* __restrict__ Chi, const __bf16* __restrict__ Clo,
    const int* __restrict__ tgt, const float* __restrict__ c2,
    const float* __restrict__ t, float* __restrict__ partial_tot,
    unsigned* __restrict__ partial_cnt) {
  __shared__ __bf16 sEhi[BM * LDSB];   // 17.0 KB
  __shared__ __bf16 sElo[BM * LDSB];   // 17.0 KB
  __shared__ __bf16 sChi[BN * LDSB];   // 68.0 KB
  __shared__ __bf16 sClo[BN * LDSB];   // 68.0 KB  (total ~170 KB of 320 KB WGP pool)
  __shared__ float red_t[8];
  __shared__ unsigned red_c[8];

  const int tid  = threadIdx.x;
  const int lane = tid & 31;
  const int wave = tid >> 5;
  const int i0   = blockIdx.y * BM;
  const int c0   = blockIdx.x * BN;

  // Stage bf16 tiles, 16B (8 x bf16) per transfer, coalesced.
#pragma unroll
  for (int r = 0; r < (BM * DIM) / (256 * 8); ++r) {  // 4 iters
    int idx = tid + r * 256;
    int row = idx >> 4;  // DIM/8 == 16 chunks per row
    int col = idx & 15;
    *reinterpret_cast<uint4*>(&sEhi[row * LDSB + col * 8]) =
        reinterpret_cast<const uint4*>(Ehi + (i0 + row) * DIM)[col];
    *reinterpret_cast<uint4*>(&sElo[row * LDSB + col * 8]) =
        reinterpret_cast<const uint4*>(Elo + (i0 + row) * DIM)[col];
  }
#pragma unroll
  for (int r = 0; r < (BN * DIM) / (256 * 8); ++r) {  // 16 iters
    int idx = tid + r * 256;
    int row = idx >> 4;
    int col = idx & 15;
    *reinterpret_cast<uint4*>(&sChi[row * LDSB + col * 8]) =
        reinterpret_cast<const uint4*>(Chi + (c0 + row) * DIM)[col];
    *reinterpret_cast<uint4*>(&sClo[row * LDSB + col * 8]) =
        reinterpret_cast<const uint4*>(Clo + (c0 + row) * DIM)[col];
  }
  __syncthreads();

  // 8 waves in a 2x4 grid; each wave owns a 32x64 subtile = 2x4 WMMA tiles.
  const int wrow  = wave >> 2;   // 0..1
  const int wcol  = wave & 3;    // 0..3
  const int hlane = lane >> 4;   // half-wave select
  const int mrow  = lane & 15;

  v8f acc[2][4] = {};
  union Frag { v16bf v; v8bf h[2]; };

#pragma unroll
  for (int k0 = 0; k0 < DIM; k0 += 32) {
    // bf16 A (16x32): lane holds K in {base..base+7} U {base+16..base+23}, base = 8*hlane
    const int abase = k0 + (hlane << 3);
    // bf16 B (32x16): lane holds K in {base..base+7} U {base+16..base+23}, base = 8*hlane
    const int bbase = k0 + (hlane << 3);
    Frag ah[2], al[2], bh[4], bl[4];
#pragma unroll
    for (int r = 0; r < 2; ++r) {
      const __bf16* ph = &sEhi[(wrow * 32 + r * 16 + mrow) * LDSB + abase];
      const __bf16* pl = &sElo[(wrow * 32 + r * 16 + mrow) * LDSB + abase];
      ah[r].h[0] = *reinterpret_cast<const v8bf*>(ph);
      ah[r].h[1] = *reinterpret_cast<const v8bf*>(ph + 16);
      al[r].h[0] = *reinterpret_cast<const v8bf*>(pl);
      al[r].h[1] = *reinterpret_cast<const v8bf*>(pl + 16);
    }
#pragma unroll
    for (int cb = 0; cb < 4; ++cb) {
      const __bf16* ph = &sChi[(wcol * 64 + cb * 16 + mrow) * LDSB + bbase];
      const __bf16* pl = &sClo[(wcol * 64 + cb * 16 + mrow) * LDSB + bbase];
      bh[cb].h[0] = *reinterpret_cast<const v8bf*>(ph);
      bh[cb].h[1] = *reinterpret_cast<const v8bf*>(ph + 16);
      bl[cb].h[0] = *reinterpret_cast<const v8bf*>(pl);
      bl[cb].h[1] = *reinterpret_cast<const v8bf*>(pl + 16);
    }
    // bf16x3: hi*hi + hi*lo + lo*hi (lo*lo dropped; ~2^-18 relative error)
#pragma unroll
    for (int r = 0; r < 2; ++r) {
#pragma unroll
      for (int cb = 0; cb < 4; ++cb) {
        acc[r][cb] = __builtin_amdgcn_wmma_f32_16x16x32_bf16(
            false, ah[r].v, false, bh[cb].v, (short)0, acc[r][cb], false, false);
        acc[r][cb] = __builtin_amdgcn_wmma_f32_16x16x32_bf16(
            false, ah[r].v, false, bl[cb].v, (short)0, acc[r][cb], false, false);
        acc[r][cb] = __builtin_amdgcn_wmma_f32_16x16x32_bf16(
            false, al[r].v, false, bh[cb].v, (short)0, acc[r][cb], false, false);
      }
    }
  }

  // Fused epilogue: loss = 1 + 2*dot - c2[col] - t[row]; mask (>0, col != tgt).
  // C/D layout: vgpr v, lane l -> M = v + 8*(l>=16), N = l & 15.
  float tot = 0.0f;
  unsigned cnt = 0u;
#pragma unroll
  for (int r = 0; r < 2; ++r) {
    const int rowbase = i0 + wrow * 32 + r * 16 + (hlane << 3);
    float t8[8];
    int tg8[8];
#pragma unroll
    for (int v = 0; v < 8; ++v) {
      t8[v]  = t[rowbase + v];
      tg8[v] = tgt[rowbase + v];
    }
#pragma unroll
    for (int cb = 0; cb < 4; ++cb) {
      const int colg = c0 + wcol * 64 + cb * 16 + mrow;
      const float c2v = c2[colg];
#pragma unroll
      for (int v = 0; v < 8; ++v) {
        float lossv = LAMBD + 2.0f * acc[r][cb][v] - c2v - t8[v];
        if (lossv > 0.0f && tg8[v] != colg) {
          tot += lossv;
          cnt += 1u;
        }
      }
    }
  }

  // Wave32 reduction, then per-workgroup partials (deterministic, no atomics).
#pragma unroll
  for (int m = 16; m >= 1; m >>= 1) {
    tot += __shfl_xor(tot, m, 32);
    cnt += __shfl_xor(cnt, m, 32);
  }
  if (lane == 0) { red_t[wave] = tot; red_c[wave] = cnt; }
  __syncthreads();
  if (tid == 0) {
    float T = 0.0f;
    unsigned Cn = 0u;
#pragma unroll
    for (int w = 0; w < 8; ++w) { T += red_t[w]; Cn += red_c[w]; }
    const int bid = blockIdx.y * gridDim.x + blockIdx.x;
    partial_tot[bid] = T;
    partial_cnt[bid] = Cn;
  }
}

// ---------------------------------------------------------------- kernel 4
__global__ __launch_bounds__(256) void final_kernel(const float* __restrict__ pt,
                                                    const unsigned* __restrict__ pc,
                                                    float* __restrict__ out) {
  __shared__ float st[256];
  __shared__ unsigned sc[256];
  float T = 0.0f;
  unsigned Cn = 0u;
  for (int j = threadIdx.x; j < NBLK; j += 256) { T += pt[j]; Cn += pc[j]; }
  st[threadIdx.x] = T;
  sc[threadIdx.x] = Cn;
  __syncthreads();
  for (int s = 128; s > 0; s >>= 1) {
    if ((int)threadIdx.x < s) {
      st[threadIdx.x] += st[threadIdx.x + s];
      sc[threadIdx.x] += sc[threadIdx.x + s];
    }
    __syncthreads();
  }
  if (threadIdx.x == 0) {
    float cntf = (float)sc[0];
    out[0] = (cntf > 0.0f) ? (st[0] / fmaxf(cntf, 1.0f)) : 0.0f;
  }
}

// ---------------------------------------------------------------- launch
extern "C" void kernel_launch(void* const* d_in, const int* in_sizes, int n_in,
                              void* d_out, int out_size, void* d_ws, size_t ws_size,
                              hipStream_t stream) {
  (void)in_sizes; (void)n_in; (void)out_size; (void)ws_size;
  const float* emb     = (const float*)d_in[0];
  const int*   tgt     = (const int*)d_in[1];
  const float* centers = (const float*)d_in[2];
  float* out = (float*)d_out;

  // Workspace layout (all fully overwritten each call -> no zero-init needed).
  float*    c2   = (float*)d_ws;                 // [4096]
  float*    t    = c2 + N_CTR;                   // [16384]
  float*    ptot = t + N_ROWS;                   // [NBLK]
  unsigned* pcnt = (unsigned*)(ptot + NBLK);     // [NBLK]
  __bf16*   Ehi  = (__bf16*)(pcnt + NBLK);       // [N_ROWS*DIM] (16B-aligned offset)
  __bf16*   Elo  = Ehi + (size_t)N_ROWS * DIM;
  __bf16*   Chi  = Elo + (size_t)N_ROWS * DIM;
  __bf16*   Clo  = Chi + (size_t)N_CTR * DIM;    // ~10.6 MB total

  split_bf16_kernel<<<(N_ROWS * DIM / 4) / 256, 256, 0, stream>>>(emb, Ehi, Elo);
  split_bf16_kernel<<<(N_CTR * DIM / 4) / 256, 256, 0, stream>>>(centers, Chi, Clo);
  c2_kernel<<<N_CTR / 8, 256, 0, stream>>>(centers, c2);
  t_kernel<<<N_ROWS / 8, 256, 0, stream>>>(emb, tgt, centers, c2, t);
  dim3 grid(N_CTR / BN, N_ROWS / BM);  // (16, 256)
  gemm_loss_kernel<<<grid, 256, 0, stream>>>(Ehi, Elo, Chi, Clo, tgt, c2, t, ptot, pcnt);
  final_kernel<<<1, 256, 0, stream>>>(ptot, pcnt, out);
}
